// Encoding_layer_33337536151579
// MI455X (gfx1250) — matrix operand order repaired and checked by reading, not compiled, not run on hardware
//
#include <hip/hip_runtime.h>
#include <math.h>

typedef __attribute__((ext_vector_type(16))) _Float16 v16h;
typedef __attribute__((ext_vector_type(8)))  _Float16 v8h;
typedef __attribute__((ext_vector_type(8)))  float    v8f;
typedef __attribute__((ext_vector_type(4)))  float    v4f;

#define NS_TOT   23152      // 2 * int(0.5 * 22050 * 1.05)
#define NS_PAD   23168      // +16 rows so tile overruns read valid memory
#define SEG_LEN  361        // NS_TOT / 64
#define NSEG     64
#define NBATCH   64
#define MFREQ    64

// Workspace layout (bytes):
//   [0, 2965504)              S[t][m]  f16, NS_PAD x 64, m contiguous
//   [2965504, 3489792)        Aw[b][n][m] f16, 64x64x64, m contiguous
//   [3489792, ...)            Wtab[64], Ptab[64] f32
#define WS_S_OFF   0
#define WS_A_OFF   ((size_t)NS_PAD * MFREQ * 2)
#define WS_T_OFF   (WS_A_OFF + (size_t)NBATCH * 4096 * 2)

// ---------------------------------------------------------------------------
// Phase 1: image -> amplitude matrix Aw[b][n][m] (f16) + freq/phase tables
// ---------------------------------------------------------------------------
__global__ __launch_bounds__(256)
void prep_kernel(const float* __restrict__ x, _Float16* __restrict__ Aw,
                 float* __restrict__ Wtab, float* __restrict__ Ptab)
{
    __shared__ float gray[64 * 64];
    __shared__ float red[256];
    const int b   = blockIdx.x;
    const int tid = threadIdx.x;

    if (b == 0 && tid == 0) {
        unsigned ir = 0u;
        for (int i = 0; i < MFREQ; ++i) {
            ir = (ir * 9301u + 49297u) % 233280u;
            Ptab[i] = (float)(6.283185307179586 * (double)ir / 233280.0);
            Wtab[i] = (float)(6.283185307179586 * 80.0 *
                              pow(95.0, (double)i / 63.0));
        }
    }

    float sum = 0.0f;
    const float* xb = x + (size_t)b * 64 * 64 * 3;
    for (int i = tid; i < 4096; i += 256) {
        const float* p = xb + i * 3;
        float g = 255.0f * (p[0] * 0.2989f + p[1] * 0.587f + p[2] * 0.114f);
        gray[i] = g;
        sum += g;
    }
    red[tid] = sum;
    __syncthreads();
    for (int s = 128; s > 0; s >>= 1) {
        if (tid < s) red[tid] += red[tid + s];
        __syncthreads();
    }
    const float avg = red[0] * (1.0f / 4096.0f);

    // px = clip(3*rev - 2*avg, 0, 255); A = px==0 ? 0 : 10^((px/16-15)/10)
    for (int i = tid; i < 4096; i += 256) {
        int n = i >> 6;                 // time column
        int m = i & 63;                 // frequency row (vertical flip)
        float rv = gray[(63 - m) * 64 + n];
        float px = 3.0f * rv - 2.0f * avg;
        px = fminf(fmaxf(px, 0.0f), 255.0f);
        float a = (px == 0.0f) ? 0.0f
                : exp2f((px * 0.0625f - 15.0f) * 0.33219280948873623f);
        Aw[(size_t)b * 4096 + i] = (_Float16)a;
    }
}

// ---------------------------------------------------------------------------
// Phase 2: materialize sin matrix S[t][m] = sin(W[m]*t*dt + PHI0[m]) as f16.
// Batch-invariant; computed once (1.48M sines), keeps libm sine out of the
// WMMA loop. Rows [NS_TOT, NS_PAD) are harmless tile-overrun padding.
// ---------------------------------------------------------------------------
__global__ __launch_bounds__(256)
void sinmat_kernel(const float* __restrict__ Wtab, const float* __restrict__ Ptab,
                   _Float16* __restrict__ S)
{
    __shared__ float sW[MFREQ];
    __shared__ float sP[MFREQ];
    const int tid = threadIdx.x;
    if (tid < MFREQ)            sW[tid] = Wtab[tid];
    else if (tid < 2 * MFREQ)   sP[tid - MFREQ] = Ptab[tid - MFREQ];
    __syncthreads();

    const float dt = 1.0f / 22050.0f;
    const int   total = NS_PAD * MFREQ;
    for (int i = blockIdx.x * 256 + tid; i < total; i += gridDim.x * 256) {
        int t = i >> 6;
        int m = i & 63;
        S[i] = (_Float16)sinf(fmaf(sW[m], (float)t * dt, sP[m]));
    }
}

// ---------------------------------------------------------------------------
// Phase 3: pure WMMA GEMM per time-column segment.
//   D[t,b] (16x16, f32) = S_tile(16t x 32m, f16) x Aseg(32m x 16b, f16), K=64
// One segment per block.x, grid.y=4 splits t-tiles, 4 waves = 4 batch slabs.
// B fragments loaded once per segment; A fragments are 4x b128 loads/tile.
// ---------------------------------------------------------------------------
__global__ __launch_bounds__(128)
void synth_kernel(const _Float16* __restrict__ Aw,
                  const _Float16* __restrict__ S,
                  float* __restrict__ out)
{
    const int tid  = threadIdx.x;
    const int wave = tid >> 5;
    const int lane = tid & 31;
    const int half = lane >> 4;                       // 0 | 1
    const int l16  = lane & 15;
    const int b0   = wave * 16;

    const int seg     = blockIdx.x;                   // time column 0..63
    const int tpart   = blockIdx.y;                   // 0..3 round-robin tiles
    const int t_begin = seg * SEG_LEN;
    const int t_end   = (seg == NSEG - 1) ? NS_TOT : t_begin + SEG_LEN;

    // B fragments: Aseg^T 32(m) x 16(b); lane<16 -> K=0..15, lane>=16 -> K=16..31.
    const _Float16* bp = Aw + (size_t)(b0 + l16) * 4096 + seg * 64 + half * 16;
    const v16h bf0 = *(const v16h*)(bp);
    const v16h bf1 = *(const v16h*)(bp + 32);

    for (int t0 = t_begin + tpart * 16; t0 < t_end; t0 += 64) {
        // A fragment row t = t0 + l16 (both lane halves map M = lane%16).
        // Per lane, elems are one contiguous run of 16 f16 per k-chunk:
        //   half=0: k 0..15 ; half=1: k 16..31.
        const _Float16* sp = S + (size_t)(t0 + l16) * MFREQ + half * 16;
        v8h a0lo = *(const v8h*)(sp);        // m = kc*32 + half*16 + 0..7
        v8h a0hi = *(const v8h*)(sp + 8);    // m = kc*32 + half*16 + 8..15
        v8h a1lo = *(const v8h*)(sp + 32);
        v8h a1hi = *(const v8h*)(sp + 40);
        __builtin_prefetch(sp + 64 * MFREQ, 0, 3);   // next tile's rows

        v16h a0 = __builtin_shufflevector(a0lo, a0hi,
                    0,1,2,3,4,5,6,7,8,9,10,11,12,13,14,15);
        v16h a1 = __builtin_shufflevector(a1lo, a1hi,
                    0,1,2,3,4,5,6,7,8,9,10,11,12,13,14,15);

        v8f c = {};
        c = __builtin_amdgcn_wmma_f32_16x16x32_f16(false, a0, false, bf0,
                                                   (short)0, c, false, false);
        c = __builtin_amdgcn_wmma_f32_16x16x32_f16(false, a1, false, bf1,
                                                   (short)0, c, false, false);

        // D: lane<16 -> b=l16, rows t0..t0+7 ; lane>=16 -> rows t0+8..15.
        const int b  = b0 + l16;
        const int tb = t0 + half * 8;
        float* op = out + (size_t)b * NS_TOT + tb;

        if (t0 + 16 <= t_end) {                       // uniform full-tile path
            v4f v0, v1;
            #pragma unroll
            for (int r = 0; r < 4; ++r) {
                float vA = 0.5f + 2048.0f * c[r];
                float vB = 0.5f + 2048.0f * c[r + 4];
                v0[r] = fminf(fmaxf(vA, -32768.0f), 32767.0f);
                v1[r] = fminf(fmaxf(vB, -32768.0f), 32767.0f);
            }
            *(v4f*)(op)     = v0;
            *(v4f*)(op + 4) = v1;
        } else {                                      // segment-boundary tile
            #pragma unroll
            for (int r = 0; r < 8; ++r) {
                if (tb + r < t_end) {
                    float v = 0.5f + 2048.0f * c[r];
                    op[r] = fminf(fmaxf(v, -32768.0f), 32767.0f);
                }
            }
        }
    }
}

// ---------------------------------------------------------------------------
extern "C" void kernel_launch(void* const* d_in, const int* in_sizes, int n_in,
                              void* d_out, int out_size, void* d_ws, size_t ws_size,
                              hipStream_t stream)
{
    (void)in_sizes; (void)n_in; (void)out_size; (void)ws_size;

    const float* x   = (const float*)d_in[0];
    float*       out = (float*)d_out;

    _Float16* S    = (_Float16*)((char*)d_ws + WS_S_OFF);
    _Float16* Aw   = (_Float16*)((char*)d_ws + WS_A_OFF);
    float*    Wtab = (float*)((char*)d_ws + WS_T_OFF);
    float*    Ptab = Wtab + MFREQ;

    prep_kernel<<<NBATCH, 256, 0, stream>>>(x, Aw, Wtab, Ptab);
    sinmat_kernel<<<1024, 256, 0, stream>>>(Wtab, Ptab, S);
    synth_kernel<<<dim3(NSEG, 4), 128, 0, stream>>>(Aw, S, out);
}
